// ProtocolTreeGAttention_63668595196274
// MI455X (gfx1250) — matrix operand order repaired and checked by reading, not compiled, not run on hardware
//
#include <hip/hip_runtime.h>
#include <hip/hip_bf16.h>

// Problem constants (from reference)
#define B_GRAPHS 2048
#define F_NODES  64
#define DIN      32
#define H_DIM    128
#define HEADS    4
#define NCLS     16
#define EPG      128
#define N_NODES  (B_GRAPHS * F_NODES)   // 131072
#define E_EDGES  (B_GRAPHS * EPG)       // 262144
#define EP_TOT   192                    // per-graph edges incl. self loops
#define H4       (HEADS * H_DIM)        // 512

typedef __attribute__((ext_vector_type(16))) __bf16        v16bf;
typedef __attribute__((ext_vector_type(8)))  float         v8f;
typedef __attribute__((ext_vector_type(8)))  unsigned int  v8u;

// Native bf16 convert (lets the compiler use gfx1250 v_cvt*bf16* ops)
__device__ __forceinline__ unsigned short f2bf(float f) {
  return __builtin_bit_cast(unsigned short, (__bf16)f);
}
__device__ __forceinline__ float bf2f(unsigned short h) {
  return __uint_as_float(((unsigned int)h) << 16);
}
// Unpack 8 bf16 (one uint4) to 8 floats: 2 VALU per pair
__device__ __forceinline__ void unpack8(uint4 q, float* o) {
  o[0] = __uint_as_float(q.x << 16); o[1] = __uint_as_float(q.x & 0xffff0000u);
  o[2] = __uint_as_float(q.y << 16); o[3] = __uint_as_float(q.y & 0xffff0000u);
  o[4] = __uint_as_float(q.z << 16); o[5] = __uint_as_float(q.z & 0xffff0000u);
  o[6] = __uint_as_float(q.w << 16); o[7] = __uint_as_float(q.w & 0xffff0000u);
}
__device__ __forceinline__ unsigned pk2(float a, float b) {
  return (unsigned)f2bf(a) | ((unsigned)f2bf(b) << 16);
}

// ---- WMMA fragment loaders (CDNA5 16-bit layouts, wave32) ----
// A: 16x32 (MxK). lane: m = lane&15 (+16*m_tile); element j: K = 16*(j>>3) + 8*(lane>>4) + (j&7)
__device__ __forceinline__ v16bf load_a_tile(const unsigned short* lds, int row_stride,
                                             int m_tile, int kk, int lane) {
  int m = m_tile * 16 + (lane & 15);
  int kbase = kk * 32 + 8 * (lane >> 4);
  const uint4* p0 = (const uint4*)(lds + m * row_stride + kbase);
  const uint4* p1 = (const uint4*)(lds + m * row_stride + kbase + 16);
  uint4 a = *p0, b = *p1;
  v8u v; v[0]=a.x; v[1]=a.y; v[2]=a.z; v[3]=a.w; v[4]=b.x; v[5]=b.y; v[6]=b.z; v[7]=b.w;
  return __builtin_bit_cast(v16bf, v);
}
// B: 32x16 (KxN), pre-packed: tile = 32 lanes x 16 bf16 contiguous (32B per lane)
__device__ __forceinline__ v16bf load_b_tile(const unsigned short* __restrict__ wp,
                                             int tile_idx, int lane) {
  const uint4* p = (const uint4*)(wp + (size_t)tile_idx * 512 + lane * 16);
  uint4 a = p[0], b = p[1];
  v8u v; v[0]=a.x; v[1]=a.y; v[2]=a.z; v[3]=a.w; v[4]=b.x; v[5]=b.y; v[6]=b.z; v[7]=b.w;
  return __builtin_bit_cast(v16bf, v);
}
// C/D: 16x16 f32: VGPR r -> M = r + 8*(lane>>4), N = lane&15. Store as bf16 to LDS.
__device__ __forceinline__ void store_c_bf16(unsigned short* lds, int row_stride,
                                             int m_tile, int n_tile, int lane, v8f c) {
  int n = n_tile * 16 + (lane & 15);
  int mbase = m_tile * 16 + 8 * (lane >> 4);
#pragma unroll
  for (int r = 0; r < 8; ++r) lds[(mbase + r) * row_stride + n] = f2bf(c[r]);
}

__device__ __forceinline__ v8f wmma_bf16(v16bf a, v16bf b, v8f c) {
  return __builtin_amdgcn_wmma_f32_16x16x32_bf16(false, a, false, b, (short)0, c, false, false);
}

// ============ Kernel 1: gate + weight pack (f32 -> bf16 WMMA tile order) ============
__global__ void k_prep(const float* __restrict__ W1, const float* __restrict__ W2,
                       const float* __restrict__ gate_logits,
                       unsigned short* __restrict__ W1p, unsigned short* __restrict__ W2p,
                       float* __restrict__ gate, float* __restrict__ out_gate) {
  int t = blockIdx.x * blockDim.x + threadIdx.x;
  if (t < 65536) {                       // W1: [128,512], tiles (kk<4, nt<32)
    int tile = t >> 9, r = t & 511;
    int L = r >> 4, j = r & 15;
    int kk = tile >> 5, nt = tile & 31;
    int k = kk * 32 + 8 * (L >> 4) + 16 * (j >> 3) + (j & 7);
    int n = nt * 16 + (L & 15);
    W1p[t] = f2bf(W1[k * H4 + n]);
  } else if (t < 131072) {               // W2: [512,128], tiles (kk<16, nt<8)
    int t2 = t - 65536;
    int tile = t2 >> 9, r = t2 & 511;
    int L = r >> 4, j = r & 15;
    int kk = tile >> 3, nt = tile & 7;
    int k = kk * 32 + 8 * (L >> 4) + 16 * (j >> 3) + (j & 7);
    int n = nt * 16 + (L & 15);
    W2p[t2] = f2bf(W2[k * H_DIM + n]);
  } else if (t < 131072 + F_NODES) {
    int i = t - 131072;
    float g = 1.0f / (1.0f + expf(-gate_logits[i]));
    gate[i] = g;
    out_gate[i] = g;                     // second tuple output
  }
}

// ============ Kernel 2: fused align + gate -> x (bf16) ============
__global__ void k_align(const float* __restrict__ emb, const float* __restrict__ W_align,
                        const float* __restrict__ b_align, const float* __restrict__ gate,
                        unsigned short* __restrict__ x) {
  __shared__ float e2[2 * DIN];
  int t = threadIdx.x;
  int n0 = blockIdx.x * 2;
  if (t < 2 * DIN) e2[t] = emb[(size_t)n0 * DIN + t];
  __syncthreads();
  int n = n0 + (t >> 7);
  int h = t & 127;
  const float* ev = &e2[(t >> 7) * DIN];
  float acc = b_align[h];
#pragma unroll
  for (int d = 0; d < DIN; ++d) acc += ev[d] * W_align[d * H_DIM + h];
  acc *= gate[n & (F_NODES - 1)];
  x[(size_t)n * H_DIM + h] = f2bf(acc);
}

// ============ Kernel 3: fully fused 2-layer GAT + mean pool, one block per graph ============
__global__ void __launch_bounds__(256, 2) k_gat(
    const unsigned short* __restrict__ x_in, const int* __restrict__ eidx,
    const unsigned short* __restrict__ W1p, const float* __restrict__ as1,
    const float* __restrict__ ad1, const float* __restrict__ b1,
    const unsigned short* __restrict__ W2p, const float* __restrict__ as2,
    const float* __restrict__ ad2, const float* __restrict__ b2,
    float* __restrict__ gout) {
  extern __shared__ char smem[];
  unsigned short* xA = (unsigned short*)smem;       // 64x128 bf16 (x; later h2) @ LDS offset 0
  unsigned short* h1 = xA + 64 * 128;               // 64x512 bf16
  unsigned short* x2 = h1 + 64 * 512;               // 64x512 bf16
  float* sS1    = (float*)(x2 + 64 * 512);          // 64x4
  float* sD1    = sS1 + 256;                        // 64x4
  float* alpha1 = sD1 + 256;                        // 192x4
  int*   esrc   = (int*)(alpha1 + 768);             // 192
  int*   etgt   = esrc + EP_TOT;                    // 192
  int*   cnt    = etgt + EP_TOT;                    // 64
  int*   off    = cnt + 64;                         // 65
  int*   cur    = off + 65;                         // 64
  int*   elist  = cur + 64;                         // 192
  float* sS2    = (float*)(elist + EP_TOT);         // 64
  float* sD2    = sS2 + 64;                         // 64
  float* alpha2 = sD2 + 64;                         // 192
  float* gpool  = alpha2 + EP_TOT;                  // 128

  const int b = blockIdx.x;
  const int tid = threadIdx.x;
  const int lane = tid & 31;
  const int wave = tid >> 5;

  // Async DMA this graph's x tile (16 KB) global -> LDS, GVS mode:
  // each lane's off16 is both its LDS byte address and its offset from the SGPR base. xA is at LDS offset 0.
  {
    unsigned long long gbase =
        (unsigned long long)(x_in + (size_t)b * F_NODES * H_DIM);
#pragma unroll
    for (int i = 0; i < 4; ++i) {
      unsigned off16 = (unsigned)(tid * 16 + i * 4096);
      asm volatile("global_load_async_to_lds_b128 %0, %1, %2"
                   :: "v"(off16), "v"(off16), "s"(gbase) : "memory");
    }
  }
  // Edges (local ids) + self loops
  if (tid < EPG) {
    esrc[tid] = eidx[b * EPG + tid] - b * F_NODES;
    etgt[tid] = eidx[E_EDGES + b * EPG + tid] - b * F_NODES;
  } else if (tid < EP_TOT) {
    esrc[tid] = tid - EPG;
    etgt[tid] = tid - EPG;
  }
  if (tid < 64) cnt[tid] = 0;
  asm volatile("s_wait_asynccnt 0" ::: "memory");   // x tile resident before barrier
  __syncthreads();
  if (tid < EP_TOT) atomicAdd(&cnt[etgt[tid]], 1);
  __syncthreads();
  if (tid == 0) {
    int s = 0;
    for (int n = 0; n < 64; ++n) { off[n] = s; s += cnt[n]; }
    off[64] = s;
  }
  __syncthreads();
  if (tid < 64) cur[tid] = off[tid];
  __syncthreads();
  if (tid < EP_TOT) { int p = atomicAdd(&cur[etgt[tid]], 1); elist[p] = tid; }
  __syncthreads();

  // -------- GEMM1: h1 = x @ W1  ([64,128]x[128,512], bf16 WMMA) --------
  {
    int mt = wave & 3;
    v16bf a0 = load_a_tile(xA, 128, mt, 0, lane);
    v16bf a1 = load_a_tile(xA, 128, mt, 1, lane);
    v16bf a2 = load_a_tile(xA, 128, mt, 2, lane);
    v16bf a3 = load_a_tile(xA, 128, mt, 3, lane);
    int ntbase = (wave >> 2) * 16;
    for (int i = 0; i < 16; ++i) {
      int nt = ntbase + i;
      v8f acc = {0.f, 0.f, 0.f, 0.f, 0.f, 0.f, 0.f, 0.f};
      acc = wmma_bf16(a0, load_b_tile(W1p, 0 * 32 + nt, lane), acc);
      acc = wmma_bf16(a1, load_b_tile(W1p, 1 * 32 + nt, lane), acc);
      acc = wmma_bf16(a2, load_b_tile(W1p, 2 * 32 + nt, lane), acc);
      acc = wmma_bf16(a3, load_b_tile(W1p, 3 * 32 + nt, lane), acc);
      store_c_bf16(h1, H4, mt, nt, lane, acc);
    }
  }
  __syncthreads();

  // -------- Attention 1: per (node, head) dots (vectorized bf16 reads) --------
  {
    int n = tid >> 2, hd = tid & 3;
    const uint4* hq = (const uint4*)&h1[n * H4 + hd * H_DIM];
    const float* av = &as1[hd * H_DIM];
    const float* dv = &ad1[hd * H_DIM];
    float ss = 0.f, sd = 0.f;
    for (int c = 0; c < 16; ++c) {
      float hv[8]; unpack8(hq[c], hv);
      int d0 = c * 8;
#pragma unroll
      for (int j = 0; j < 8; ++j) { ss += hv[j] * av[d0 + j]; sd += hv[j] * dv[d0 + j]; }
    }
    sS1[n * 4 + hd] = ss; sD1[n * 4 + hd] = sd;
  }
  __syncthreads();
  // -------- Segment softmax 1: thread = (tgt, head) --------
  {
    int n = tid >> 2, hd = tid & 3;
    int s0 = off[n], s1 = off[n + 1];
    float sdv = sD1[n * 4 + hd];
    float m = -3.0e38f;
    for (int k = s0; k < s1; ++k) {
      float sc = sS1[esrc[elist[k]] * 4 + hd] + sdv;
      sc = sc > 0.f ? sc : 0.2f * sc;
      m = fmaxf(m, sc);
    }
    float sum = 0.f;
    for (int k = s0; k < s1; ++k) {
      float sc = sS1[esrc[elist[k]] * 4 + hd] + sdv;
      sc = sc > 0.f ? sc : 0.2f * sc;
      sum += expf(sc - m);
    }
    float inv = 1.0f / (sum + 1e-16f);
    for (int k = s0; k < s1; ++k) {
      int e = elist[k];
      float sc = sS1[esrc[e] * 4 + hd] + sdv;
      sc = sc > 0.f ? sc : 0.2f * sc;
      alpha1[e * 4 + hd] = expf(sc - m) * inv;
    }
  }
  __syncthreads();
  // -------- Aggregation 1 + bias + ELU -> x2 (vector LDS loads/stores) --------
  {
    int n = tid >> 2, q = tid & 3;     // q == head (128 feats each)
    int s0 = off[n], s1 = off[n + 1];
    for (int c = 0; c < 16; ++c) {
      int f0 = q * 128 + c * 8;
      float acc[8] = {0, 0, 0, 0, 0, 0, 0, 0};
      for (int k = s0; k < s1; ++k) {
        int e = elist[k];
        float a = alpha1[e * 4 + q];
        float hv[8]; unpack8(*(const uint4*)&h1[esrc[e] * H4 + f0], hv);
#pragma unroll
        for (int j = 0; j < 8; ++j) acc[j] += a * hv[j];
      }
      float v[8];
#pragma unroll
      for (int j = 0; j < 8; ++j) {
        float t = acc[j] + b1[f0 + j];
        v[j] = t > 0.f ? t : (expf(t) - 1.0f);   // ELU
      }
      uint4 pk = { pk2(v[0], v[1]), pk2(v[2], v[3]), pk2(v[4], v[5]), pk2(v[6], v[7]) };
      *(uint4*)&x2[n * H4 + f0] = pk;
    }
  }
  __syncthreads();

  // -------- GEMM2: h2 = x2 @ W2  ([64,512]x[512,128]) -> xA (reuse) --------
  {
    int mt = wave & 3;
    int ntbase = (wave >> 2) * 4;
    v8f acc[4];
#pragma unroll
    for (int i = 0; i < 4; ++i) acc[i] = (v8f){0.f, 0.f, 0.f, 0.f, 0.f, 0.f, 0.f, 0.f};
    for (int kk = 0; kk < 16; ++kk) {
      v16bf a = load_a_tile(x2, H4, mt, kk, lane);
#pragma unroll
      for (int i = 0; i < 4; ++i)
        acc[i] = wmma_bf16(a, load_b_tile(W2p, kk * 8 + ntbase + i, lane), acc[i]);
    }
#pragma unroll
    for (int i = 0; i < 4; ++i) store_c_bf16(xA, H_DIM, mt, ntbase + i, lane, acc[i]);
  }
  __syncthreads();

  // -------- Attention 2 (1 head) --------
  if (tid < 64) {
    int n = tid;
    const uint4* hq = (const uint4*)&xA[n * H_DIM];
    float ss = 0.f, sd = 0.f;
    for (int c = 0; c < 16; ++c) {
      float hv[8]; unpack8(hq[c], hv);
      int d0 = c * 8;
#pragma unroll
      for (int j = 0; j < 8; ++j) { ss += hv[j] * as2[d0 + j]; sd += hv[j] * ad2[d0 + j]; }
    }
    sS2[n] = ss; sD2[n] = sd;
  }
  if (tid < 128) gpool[tid] = 0.f;
  __syncthreads();
  if (tid < 64) {
    int n = tid;
    int s0 = off[n], s1 = off[n + 1];
    float sdv = sD2[n];
    float m = -3.0e38f;
    for (int k = s0; k < s1; ++k) {
      float sc = sS2[esrc[elist[k]]] + sdv;
      sc = sc > 0.f ? sc : 0.2f * sc;
      m = fmaxf(m, sc);
    }
    float sum = 0.f;
    for (int k = s0; k < s1; ++k) {
      float sc = sS2[esrc[elist[k]]] + sdv;
      sc = sc > 0.f ? sc : 0.2f * sc;
      sum += expf(sc - m);
    }
    float inv = 1.0f / (sum + 1e-16f);
    for (int k = s0; k < s1; ++k) {
      int e = elist[k];
      float sc = sS2[esrc[e]] + sdv;
      sc = sc > 0.f ? sc : 0.2f * sc;
      alpha2[e] = expf(sc - m) * inv;
    }
  }
  __syncthreads();
  // -------- Aggregation 2 + bias + mean pool --------
  {
    int n = tid >> 2, q = tid & 3;
    int s0 = off[n], s1 = off[n + 1];
    for (int c = 0; c < 4; ++c) {
      int d0 = q * 32 + c * 8;
      float acc[8] = {0, 0, 0, 0, 0, 0, 0, 0};
      for (int k = s0; k < s1; ++k) {
        int e = elist[k];
        float a = alpha2[e];
        float hv[8]; unpack8(*(const uint4*)&xA[esrc[e] * H_DIM + d0], hv);
#pragma unroll
        for (int j = 0; j < 8; ++j) acc[j] += a * hv[j];
      }
#pragma unroll
      for (int j = 0; j < 8; ++j) atomicAdd(&gpool[d0 + j], acc[j] + b2[d0 + j]);
    }
  }
  __syncthreads();
  if (tid < 128) gout[(size_t)b * H_DIM + tid] = gpool[tid] * (1.0f / 64.0f);
}

// ============ Kernel 4: classifier MLP ============
__global__ void k_cls(const float* __restrict__ g, const float* __restrict__ Wc1,
                      const float* __restrict__ bc1, const float* __restrict__ Wc2,
                      const float* __restrict__ bc2, float* __restrict__ out) {
  __shared__ float gg[128];
  __shared__ float hc[64];
  int b = blockIdx.x, t = threadIdx.x;
  gg[t] = g[(size_t)b * 128 + t];
  gg[t + 64] = g[(size_t)b * 128 + 64 + t];
  __syncthreads();
  float a = bc1[t];
  for (int d = 0; d < 128; ++d) a += gg[d] * Wc1[d * 64 + t];
  hc[t] = a > 0.f ? a : 0.01f * a;
  __syncthreads();
  if (t < NCLS) {
    float o = bc2[t];
    for (int k = 0; k < 64; ++k) o += hc[k] * Wc2[k * NCLS + t];
    out[(size_t)b * NCLS + t] = o;
  }
}

extern "C" void kernel_launch(void* const* d_in, const int* in_sizes, int n_in,
                              void* d_out, int out_size, void* d_ws, size_t ws_size,
                              hipStream_t stream) {
  (void)in_sizes; (void)n_in; (void)out_size; (void)ws_size;
  const float* emb         = (const float*)d_in[0];
  const int*   eidx        = (const int*)d_in[1];
  /* d_in[2] batch_idx unused: graphs are structurally disjoint (64 nodes each) */
  const float* W_align     = (const float*)d_in[3];
  const float* b_align     = (const float*)d_in[4];
  const float* gate_logits = (const float*)d_in[5];
  const float* W1  = (const float*)d_in[6];
  const float* as1 = (const float*)d_in[7];
  const float* ad1 = (const float*)d_in[8];
  const float* b1  = (const float*)d_in[9];
  const float* W2  = (const float*)d_in[10];
  const float* as2 = (const float*)d_in[11];
  const float* ad2 = (const float*)d_in[12];
  const float* b2  = (const float*)d_in[13];
  const float* Wc1 = (const float*)d_in[14];
  const float* bc1 = (const float*)d_in[15];
  const float* Wc2 = (const float*)d_in[16];
  const float* bc2 = (const float*)d_in[17];
  float* out = (float*)d_out;

  // Workspace layout (bytes)
  char* ws = (char*)d_ws;
  unsigned short* x   = (unsigned short*)(ws + 0);          // N*128 bf16 = 33,554,432 B
  unsigned short* W1p = (unsigned short*)(ws + 33554432);   // 131,072 B
  unsigned short* W2p = (unsigned short*)(ws + 33685504);   // 131,072 B
  float*          gate = (float*)(ws + 33816576);           // 256 B
  float*          g    = (float*)(ws + 33816832);           // 2048*128 f32 = 1,048,576 B

  k_prep<<<(131072 + F_NODES + 255) / 256, 256, 0, stream>>>(
      W1, W2, gate_logits, W1p, W2p, gate, out + (size_t)B_GRAPHS * NCLS);
  k_align<<<N_NODES / 2, 256, 0, stream>>>(emb, W_align, b_align, gate, x);

  // Dynamic LDS: bf16 region 147456 B + f32/i32 softmax/CSR state 9988 B
  size_t smem = (size_t)(64 * 128 + 64 * 512 + 64 * 512) * 2
              + (size_t)(256 + 256 + 768) * 4
              + (size_t)(192 + 192 + 64 + 65 + 64 + 192) * 4
              + (size_t)(64 + 64 + 192 + 128) * 4;          // = 157,444 B (2 blocks/WGP)
  k_gat<<<B_GRAPHS, 256, smem, stream>>>(x, eidx, W1p, as1, ad1, b1, W2p, as2, ad2, b2, g);

  k_cls<<<B_GRAPHS, 64, 0, stream>>>(g, Wc1, bc1, Wc2, bc2, out);
}
